// model_bv_masked_41841571397941
// MI455X (gfx1250) — 8-bit matrix operand order repaired and checked by reading, compiled, not run on hardware
//
#include <hip/hip_runtime.h>
#include <hip/hip_bf16.h>
#include <stdint.h>


typedef __attribute__((ext_vector_type(8))) int v8i;

#define BATCH   16384
#define DIM     1024
#define OUT_N   10
#define OUT_PAD 16

#define WS_SCAL   0
#define WS_BIASI  4096
#define WS_WQ     32768
#define WS_WQ7    6324224
#define WS_ACT0   6340608
#define WS_ACT1   23117824
#define WS_ACC    39895040


__global__ void k_init_scal(unsigned* scal) {
    if (threadIdx.x < 64) scal[threadIdx.x] = 0u;
}

__global__ void k_absmax(const float* __restrict__ x, int n, unsigned* slot) {
    __shared__ unsigned sm[256];
    unsigned v = 0u;
    for (int i = blockIdx.x * blockDim.x + threadIdx.x; i < n; i += gridDim.x * blockDim.x)
        v = max(v, __float_as_uint(fabsf(x[i])));
    sm[threadIdx.x] = v; __syncthreads();
    for (int s = 128; s > 0; s >>= 1) {
        if (threadIdx.x < s) sm[threadIdx.x] = max(sm[threadIdx.x], sm[threadIdx.x + s]);
        __syncthreads();
    }
    if (threadIdx.x == 0) atomicMax(slot, sm[0]);
}

__global__ void k_wabsmax(const float* __restrict__ w, const float* __restrict__ m,
                          int n, unsigned* slot) {
    __shared__ unsigned sm[256];
    unsigned v = 0u;
    for (int i = blockIdx.x * blockDim.x + threadIdx.x; i < n; i += gridDim.x * blockDim.x)
        v = max(v, __float_as_uint(fabsf(w[i] * m[i])));
    sm[threadIdx.x] = v; __syncthreads();
    for (int s = 128; s > 0; s >>= 1) {
        if (threadIdx.x < s) sm[threadIdx.x] = max(sm[threadIdx.x], sm[threadIdx.x + s]);
        __syncthreads();
    }
    if (threadIdx.x == 0) atomicMax(slot, sm[0]);
}

__global__ void k_quant_x(const float* __restrict__ x, signed char* __restrict__ out,
                          const unsigned* __restrict__ scal, int n) {
    const float inv = 7.0f / fmaxf(__uint_as_float(scal[0]), 1e-8f);
    for (int i = blockIdx.x * blockDim.x + threadIdx.x; i < n; i += gridDim.x * blockDim.x) {
        int q = (int)rintf(x[i] * inv);
        q = q < -8 ? -8 : (q > 7 ? 7 : q);
        out[i] = (signed char)q;
    }
}

__global__ void k_quant_w(const float* __restrict__ w, const float* __restrict__ m,
                          signed char* __restrict__ out, const unsigned* __restrict__ scal,
                          int layer, int n) {
    const float inv = 7.0f / fmaxf(__uint_as_float(scal[layer]), 1e-8f);
    for (int i = blockIdx.x * blockDim.x + threadIdx.x; i < n; i += gridDim.x * blockDim.x) {
        int q = (int)rintf(w[i] * m[i] * inv);
        q = q < -7 ? -7 : (q > 7 ? 7 : q);
        out[i] = (signed char)q;
    }
}

__global__ void k_quant_w7(const float* __restrict__ w, const float* __restrict__ m,
                           signed char* __restrict__ out, const unsigned* __restrict__ scal) {
    const float inv = 7.0f / fmaxf(__uint_as_float(scal[7]), 1e-8f);
    for (int i = blockIdx.x * blockDim.x + threadIdx.x; i < OUT_PAD * DIM;
         i += gridDim.x * blockDim.x) {
        const int o = i >> 10;
        int q = 0;
        if (o < OUT_N) {
            q = (int)rintf(w[i] * m[i] * inv);
            q = q < -7 ? -7 : (q > 7 ? 7 : q);
        }
        out[i] = (signed char)q;
    }
}

__global__ void k_bias(const float* __restrict__ b, unsigned* scal,
                       int* __restrict__ bi, int layer, int o_real) {
    float* scalf = (float*)scal;
    const float wsm = fmaxf(__uint_as_float(scal[layer]), 1e-8f) * (1.0f / 7.0f);
    float sprev;
    if (layer == 1) {
        sprev = fmaxf(__uint_as_float(scal[0]), 1e-8f) * (1.0f / 7.0f);
    } else {
        const float bsp = scalf[15 + layer - 1];
        const int   mip = (int)scal[7 + layer - 1];
        sprev = fmaxf((float)mip * bsp, 1e-8f) * (1.0f / 15.0f);
    }
    const float bs = sprev * wsm;
    if (threadIdx.x == 0) scalf[15 + layer] = bs;
    const int o = threadIdx.x;
    int q = 0;
    if (o < o_real) {
        q = (int)rintf(b[o] / bs);
        q = q < -128 ? -128 : (q > 127 ? 127 : q);
    }
    bi[o] = q;
}

__device__ __forceinline__ unsigned lds_off(const void* p) {
    return (unsigned)(unsigned long long)p;
}

#define ASYNC_LDS_B128(ldst, gsrc) \
    asm volatile("global_load_async_to_lds_b128 %0, %1, off" \
                 :: "v"(ldst), "v"(gsrc) : "memory")

template<bool SIGNED_A>
__global__ __launch_bounds__(256) void k_gemm(const signed char* __restrict__ A,
                                              const signed char* __restrict__ W,
                                              const int* __restrict__ bi,
                                              int* __restrict__ C,
                                              int* __restrict__ maxslot) {
    constexpr int KD = DIM, LSTR = 96, KSTEPS = KD / 64;
    constexpr unsigned BUFSZ = 128 * LSTR;
    __shared__ char sA[2 * 128 * LSTR];
    __shared__ char sW[2 * 128 * LSTR];
    __shared__ int  blkmax;

    const int tid  = threadIdx.x;
    const int lane = tid & 31;
    const int wv   = tid >> 5;
    const int wr   = wv & 3;
    const int wc   = wv >> 2;
    const int mbase = blockIdx.x * 128;
    const int nbase = blockIdx.y * 128;
    const int lrow   = tid >> 2;
    const int lchunk = (tid & 3) * 16;
    const int l16  = lane & 15;
    const int half = lane >> 4;

    if (tid == 0) blkmax = 0;

    const unsigned long long gA0 = (unsigned long long)(A + (size_t)(mbase + lrow)      * KD + lchunk);
    const unsigned long long gA1 = (unsigned long long)(A + (size_t)(mbase + lrow + 64) * KD + lchunk);
    const unsigned long long gW0 = (unsigned long long)(W + (size_t)(nbase + lrow)      * KD + lchunk);
    const unsigned long long gW1 = (unsigned long long)(W + (size_t)(nbase + lrow + 64) * KD + lchunk);
    const unsigned dA0 = lds_off(&sA[lrow * LSTR + lchunk]);
    const unsigned dA1 = dA0 + 64u * LSTR;
    const unsigned dW0 = lds_off(&sW[lrow * LSTR + lchunk]);
    const unsigned dW1 = dW0 + 64u * LSTR;

    ASYNC_LDS_B128(dA0, gA0);
    ASYNC_LDS_B128(dA1, gA1);
    ASYNC_LDS_B128(dW0, gW0);
    ASYNC_LDS_B128(dW1, gW1);

    v8i acc[2][4];
#pragma unroll
    for (int i = 0; i < 2; i++)
#pragma unroll
        for (int j = 0; j < 4; j++) acc[i][j] = v8i{0, 0, 0, 0, 0, 0, 0, 0};

    for (int kt = 0; kt < KSTEPS; kt++) {
        asm volatile("s_wait_asynccnt 0" ::: "memory");
        __syncthreads();

        if (kt + 1 < KSTEPS) {
            const unsigned nb   = (unsigned)((kt + 1) & 1) * BUFSZ;
            const unsigned long long ko = (unsigned long long)(kt + 1) * 64ull;
            ASYNC_LDS_B128(dA0 + nb, gA0 + ko);
            ASYNC_LDS_B128(dA1 + nb, gA1 + ko);
            ASYNC_LDS_B128(dW0 + nb, gW0 + ko);
            ASYNC_LDS_B128(dW1 + nb, gW1 + ko);
        }

        const unsigned cb = (unsigned)(kt & 1) * BUFSZ;

        v8i af[2];
#pragma unroll
        for (int ms = 0; ms < 2; ms++) {
            const char* rp = &sA[cb + (unsigned)(wr * 32 + ms * 16 + l16) * LSTR];
#pragma unroll
            for (int c = 0; c < 2; c++) {
                const uint4 u = *(const uint4*)(rp + c * 32 + half * 16);
                af[ms][4 * c]     = (int)u.x;
                af[ms][4 * c + 1] = (int)u.y;
                af[ms][4 * c + 2] = (int)u.z;
                af[ms][4 * c + 3] = (int)u.w;
            }
        }
#pragma unroll
        for (int ns = 0; ns < 4; ns++) {
            const char* rp = &sW[cb + (unsigned)(wc * 64 + ns * 16 + l16) * LSTR];
            const uint4 lo = *(const uint4*)(rp + half * 16);
            const uint4 hi = *(const uint4*)(rp + 32 + half * 16);
            v8i bf;
            bf[0] = (int)lo.x; bf[1] = (int)lo.y; bf[2] = (int)lo.z; bf[3] = (int)lo.w;
            bf[4] = (int)hi.x; bf[5] = (int)hi.y; bf[6] = (int)hi.z; bf[7] = (int)hi.w;
#pragma unroll
            for (int ms = 0; ms < 2; ms++)
                acc[ms][ns] = __builtin_amdgcn_wmma_i32_16x16x64_iu8(
                    SIGNED_A, af[ms], true, bf, acc[ms][ns], false, false);
        }
    }

    int lmax = 0;
#pragma unroll
    for (int ms = 0; ms < 2; ms++) {
#pragma unroll
        for (int ns = 0; ns < 4; ns++) {
            const int col = nbase + wc * 64 + ns * 16 + l16;
            const int bv  = bi[col];
            const int r0  = mbase + wr * 32 + ms * 16 + half * 8;
#pragma unroll
            for (int r = 0; r < 8; r++) {
                const int v = acc[ms][ns][r] + bv;
                lmax = v > lmax ? v : lmax;
                C[(size_t)(r0 + r) * DIM + col] = v;
            }
        }
    }
    atomicMax(&blkmax, lmax);
    __syncthreads();
    if (tid == 0) atomicMax(maxslot, blkmax);
}

__global__ void k_requant(const int* __restrict__ acc, signed char* __restrict__ out,
                          const unsigned* __restrict__ scal, int layer, int n) {
    const float bs = ((const float*)scal)[15 + layer];
    const int   mi = (int)scal[7 + layer];
    const float f  = bs * 15.0f / fmaxf((float)mi * bs, 1e-8f);
    for (int i = blockIdx.x * blockDim.x + threadIdx.x; i < n; i += gridDim.x * blockDim.x) {
        int a = acc[i];
        if (a < 0) a = 0;
        int q = (int)rintf((float)a * f);
        q = q < 0 ? 0 : (q > 15 ? 15 : q);
        out[i] = (signed char)q;
    }
}

__global__ __launch_bounds__(256) void k_gemm7(const signed char* __restrict__ A,
                                               const signed char* __restrict__ W16,
                                               const int* __restrict__ bi,
                                               const unsigned* __restrict__ scal,
                                               float* __restrict__ out) {
    const int tid  = threadIdx.x;
    const int lane = tid & 31;
    const int wv   = tid >> 5;
    const int rowbase = (blockIdx.x * 8 + wv) * 16;
    const int l16  = lane & 15;
    const int half = lane >> 4;
    const float bs7 = ((const float*)scal)[15 + 7];

    v8i acc = {0, 0, 0, 0, 0, 0, 0, 0};
    const signed char* arow = A   + (size_t)(rowbase + l16) * DIM;
    const signed char* wrow = W16 + (size_t)l16 * DIM;

    for (int k0 = 0; k0 < DIM; k0 += 64) {
        v8i af;
#pragma unroll
        for (int c = 0; c < 2; c++) {
            const uint4 u = *(const uint4*)(arow + k0 + c * 32 + half * 16);
            af[4 * c]     = (int)u.x;
            af[4 * c + 1] = (int)u.y;
            af[4 * c + 2] = (int)u.z;
            af[4 * c + 3] = (int)u.w;
        }
        const uint4 lo = *(const uint4*)(wrow + k0 + half * 16);
        const uint4 hi = *(const uint4*)(wrow + k0 + 32 + half * 16);
        v8i bf;
        bf[0] = (int)lo.x; bf[1] = (int)lo.y; bf[2] = (int)lo.z; bf[3] = (int)lo.w;
        bf[4] = (int)hi.x; bf[5] = (int)hi.y; bf[6] = (int)hi.z; bf[7] = (int)hi.w;
        acc = __builtin_amdgcn_wmma_i32_16x16x64_iu8(false, af, true, bf, acc, false, false);
    }

    if (l16 < OUT_N) {
        const int bv = bi[l16];
#pragma unroll
        for (int r = 0; r < 8; r++) {
            const int row = rowbase + r + half * 8;
            out[(size_t)row * OUT_N + l16] = (float)(acc[r] + bv) * bs7;
        }
    }
}

extern "C" void kernel_launch(void* const* d_in, const int* in_sizes, int n_in,
                              void* d_out, int out_size, void* d_ws, size_t ws_size,
                              hipStream_t stream) {
    (void)in_sizes; (void)n_in; (void)out_size; (void)ws_size;
    const float* x = (const float*)d_in[0];
    const float* w[8]; const float* b[8]; const float* m[8];
    for (int l = 1; l <= 7; l++) {
        w[l] = (const float*)d_in[3 * l - 2];
        b[l] = (const float*)d_in[3 * l - 1];
        m[l] = (const float*)d_in[3 * l];
    }

    char* ws = (char*)d_ws;
    unsigned*    scal  = (unsigned*)(ws + WS_SCAL);
    int*         biasi = (int*)(ws + WS_BIASI);
    signed char* wq[8];
    for (int l = 1; l <= 6; l++) wq[l] = (signed char*)(ws + WS_WQ + (size_t)(l - 1) * DIM * DIM);
    wq[7] = (signed char*)(ws + WS_WQ7);
    signed char* act0 = (signed char*)(ws + WS_ACT0);
    signed char* act1 = (signed char*)(ws + WS_ACT1);
    int*         acc  = (int*)(ws + WS_ACC);
    float*       out  = (float*)d_out;

    const int nAct = BATCH * DIM;
    const int nW   = DIM * DIM;

    hipLaunchKernelGGL(k_init_scal, dim3(1), dim3(64), 0, stream, scal);
    hipLaunchKernelGGL(k_absmax, dim3(1024), dim3(256), 0, stream, x, nAct, &scal[0]);
    for (int l = 1; l <= 6; l++)
        hipLaunchKernelGGL(k_wabsmax, dim3(512), dim3(256), 0, stream, w[l], m[l], nW, &scal[l]);
    hipLaunchKernelGGL(k_wabsmax, dim3(40), dim3(256), 0, stream, w[7], m[7], OUT_N * DIM, &scal[7]);
    hipLaunchKernelGGL(k_quant_x, dim3(2048), dim3(256), 0, stream, x, act0, scal, nAct);
    for (int l = 1; l <= 6; l++)
        hipLaunchKernelGGL(k_quant_w, dim3(1024), dim3(256), 0, stream, w[l], m[l], wq[l], scal, l, nW);
    hipLaunchKernelGGL(k_quant_w7, dim3(64), dim3(256), 0, stream, w[7], m[7], wq[7], scal);

    signed char* src = act0;
    signed char* dst = act1;
    for (int l = 1; l <= 6; l++) {
        hipLaunchKernelGGL(k_bias, dim3(1), dim3(DIM), 0, stream,
                           b[l], scal, biasi + (l - 1) * DIM, l, DIM);
        if (l == 1)
            hipLaunchKernelGGL((k_gemm<true>), dim3(BATCH / 128, DIM / 128), dim3(256), 0, stream,
                               src, wq[l], biasi + (l - 1) * DIM, acc, (int*)&scal[7 + l]);
        else
            hipLaunchKernelGGL((k_gemm<false>), dim3(BATCH / 128, DIM / 128), dim3(256), 0, stream,
                               src, wq[l], biasi + (l - 1) * DIM, acc, (int*)&scal[7 + l]);
        hipLaunchKernelGGL(k_requant, dim3(2048), dim3(256), 0, stream, acc, dst, scal, l, nAct);
        signed char* t = src; src = dst; dst = t;
    }

    hipLaunchKernelGGL(k_bias, dim3(1), dim3(OUT_PAD), 0, stream,
                       b[7], scal, biasi + 6 * DIM, 7, OUT_N);
    hipLaunchKernelGGL(k_gemm7, dim3(BATCH / 128), dim3(256), 0, stream,
                       src, wq[7], biasi + 6 * DIM, scal, out);
}
